// PREEP_56384330662496
// MI455X (gfx1250) — compile-verified, matrix operand order checked by reading, not run on hardware
//
#include <hip/hip_runtime.h>
#include <hip/hip_bf16.h>

typedef __attribute__((ext_vector_type(16))) __bf16 v16bf;
typedef __attribute__((ext_vector_type(8)))  __bf16 v8bf;
typedef __attribute__((ext_vector_type(8)))  float  v8f;

#define BN_EPS 1e-5f

__device__ __forceinline__ __bf16 f2bf(float f) {
  unsigned u = __float_as_uint(f);
  unsigned r = (u + 0x7FFFu + ((u >> 16) & 1u)) >> 16;   // round-to-nearest-even
  unsigned short s = (unsigned short)r;
  __bf16 b; __builtin_memcpy(&b, &s, 2); return b;
}

// Combine two contiguous 8-elem halves (k..k+7, k+16..k+23) into a WMMA fragment
__device__ __forceinline__ v16bf load_a_frag(const __bf16* p) {
  v8bf lo = *(const v8bf*)p;
  v8bf hi = *(const v8bf*)(p + 16);
  return __builtin_shufflevector(lo, hi, 0, 1, 2, 3, 4, 5, 6, 7,
                                 8, 9, 10, 11, 12, 13, 14, 15);
}

// ---------------- layout shuffles ----------------
__global__ void k_nchw_to_pix(const float* __restrict__ src, float* __restrict__ dst, int n) {
  int i = blockIdx.x * blockDim.x + threadIdx.x;
  if (i >= n) return;                       // n = B*512*64, src [b][c][p]
  int b = i / (512 * 64); int rem = i - b * 512 * 64;
  int c = rem / 64; int p = rem - c * 64;
  dst[((size_t)b * 64 + p) * 512 + c] = src[i];
}
__global__ void k_pix_to_nchw(const float* __restrict__ src, float* __restrict__ dst, int n) {
  int i = blockIdx.x * blockDim.x + threadIdx.x;
  if (i >= n) return;
  int b = i / (512 * 64); int rem = i - b * 512 * 64;
  int c = rem / 64; int p = rem - c * 64;
  dst[i] = src[((size_t)b * 64 + p) * 512 + c];
}
__global__ void k_cast_bf16(const float* __restrict__ src, __bf16* __restrict__ dst, int n) {
  int i = blockIdx.x * blockDim.x + threadIdx.x;
  if (i < n) dst[i] = f2bf(src[i]);
}

// fp32 row-major W[K][N] -> WMMA-fragment-major bf16: [(n/16)][(K/32)][lane32][16]
__global__ void k_pack_b_frag(const float* __restrict__ src, __bf16* __restrict__ dst,
                              int N, int K, int n) {
  int i = blockIdx.x * blockDim.x + threadIdx.x;
  if (i >= n) return;
  int j = i & 15; int t = i >> 4;
  int l = t & 31; t >>= 5;
  int kb = t % (K >> 5); int tn = t / (K >> 5);
  int nn = (tn << 4) + (l & 15);
  int halo = (l >> 4) << 3;
  int kk = (kb << 5) + (j & 7) + halo + ((j >> 3) << 4);
  dst[i] = f2bf(src[(size_t)kk * N + nn]);
}

// conv weights [E,O,I,3,3] fp32 -> per-expert fragment-major bf16 (N=O, K=I*9)
__global__ void k_pack_conv_frag(const float* __restrict__ src, __bf16* __restrict__ dst,
                                 int O, int I, int n) {
  int i = blockIdx.x * blockDim.x + threadIdx.x;
  if (i >= n) return;
  int K = I * 9;
  int j = i & 15; int t = i >> 4;
  int l = t & 31; t >>= 5;
  int kb = t % (K >> 5); t /= (K >> 5);
  int tn = t % (O >> 4); int e = t / (O >> 4);
  int nn = (tn << 4) + (l & 15);
  int halo = (l >> 4) << 3;
  int kk = (kb << 5) + (j & 7) + halo + ((j >> 3) << 4);
  int c = kk / 9; int r = kk - c * 9;
  dst[i] = f2bf(src[(((size_t)e * O + nn) * I + c) * 9 + r]);
}

// im2col with SAME padding: pixel-major [B][64][Cin] (fp32) -> [B*64][Cin*9] bf16
__global__ void k_im2col_f32(const float* __restrict__ src, __bf16* __restrict__ dst,
                             int Cin, int n) {
  int i = blockIdx.x * blockDim.x + threadIdx.x;
  if (i >= n) return;
  int K = Cin * 9;
  int bp = i / K; int k = i - bp * K;
  int c = k / 9; int r = k - c * 9;
  int b = bp >> 6; int p = bp & 63;
  int iy = (p >> 3) + r / 3 - 1;
  int ix = (p & 7) + (r % 3) - 1;
  float v = ((iy >= 0) & (iy < 8) & (ix >= 0) & (ix < 8))
                ? src[((size_t)b * 64 + iy * 8 + ix) * Cin + c] : 0.f;
  dst[i] = f2bf(v);
}
__global__ void k_im2col_bf16(const __bf16* __restrict__ src, __bf16* __restrict__ dst,
                              int Cin, int n) {
  int i = blockIdx.x * blockDim.x + threadIdx.x;
  if (i >= n) return;
  int K = Cin * 9;
  int bp = i / K; int k = i - bp * K;
  int c = k / 9; int r = k - c * 9;
  int b = bp >> 6; int p = bp & 63;
  int iy = (p >> 3) + r / 3 - 1;
  int ix = (p & 7) + (r % 3) - 1;
  __bf16 z = f2bf(0.f);
  dst[i] = ((iy >= 0) & (iy < 8) & (ix >= 0) & (ix < 8))
               ? src[((size_t)b * 64 + iy * 8 + ix) * Cin + c] : z;
}

// ---------------- gate: pooled features -> top-1 expert ----------------
__global__ void k_gate(const float* __restrict__ xp, const float* __restrict__ gw1,
                       const float* __restrict__ gb1, const float* __restrict__ gw2,
                       const float* __restrict__ gb2, int E, int* __restrict__ idx) {
  __shared__ float g[512];
  __shared__ float t[256];
  int b = blockIdx.x, tid = threadIdx.x;
  for (int c = tid; c < 512; c += 256) {
    float s = 0.f;
    for (int p = 0; p < 64; ++p) s += xp[((size_t)b * 64 + p) * 512 + c];
    g[c] = s * (1.f / 64.f);
  }
  __syncthreads();
  {
    float s = gb1[tid];
    for (int c = 0; c < 512; ++c) s += g[c] * gw1[c * 256 + tid];
    t[tid] = fmaxf(s, 0.f);
  }
  __syncthreads();
  if (tid == 0) {
    int best = 0; float bv = -3.4e38f;
    for (int e = 0; e < E; ++e) {
      float s = gb2[e];
      for (int h = 0; h < 256; ++h) s += t[h] * gw2[h * E + e];
      if (s > bv) { bv = s; best = e; }   // argmax(softmax)==argmax(logits)
    }
    idx[b] = best;
  }
}

// -------- bf16 WMMA GEMM, fragment-packed B, 16x32 output per wave --------
template <bool RELU, bool STORE_F32, bool STORE_BF16>
__global__ __launch_bounds__(256)
void k_gemm_wmma(const __bf16* __restrict__ A, const __bf16* __restrict__ Bp,
                 const float* __restrict__ bias, float* __restrict__ Cf,
                 __bf16* __restrict__ Cb, int M, int N, int K) {
  int lane = threadIdx.x & 31;
  int wave = threadIdx.x >> 5;
  int tilesN2 = N >> 5;                      // wave covers two 16-wide N tiles
  int wt = blockIdx.x * 8 + wave;            // wave-uniform
  int tmIdx = wt / tilesN2;
  int tn2 = wt - tmIdx * tilesN2;
  int tm = tmIdx << 4;
  if (tm >= M) return;                        // uniform across wave
  int kblocks = K >> 5;
  int halo = (lane >> 4) << 3;
  const __bf16* Arow = A + (size_t)(tm + (lane & 15)) * K + halo;
  const __bf16* B0 = Bp + ((size_t)(2 * tn2) * kblocks * 32 + lane) * 16;
  const __bf16* B1 = B0 + (size_t)kblocks * 32 * 16;
  v8f acc0 = {}, acc1 = {};
  for (int kb = 0; kb < kblocks; ++kb) {
    __builtin_prefetch(Arow + ((kb + 4) << 5), 0, 0);
    __builtin_prefetch(B0 + (size_t)(kb + 4) * 512, 0, 0);
    __builtin_prefetch(B1 + (size_t)(kb + 4) * 512, 0, 0);
    v16bf av = load_a_frag(Arow + (kb << 5));
    v16bf b0 = *(const v16bf*)(B0 + (size_t)kb * 512);
    v16bf b1 = *(const v16bf*)(B1 + (size_t)kb * 512);
    acc0 = __builtin_amdgcn_wmma_f32_16x16x32_bf16(false, av, false, b0,
                                                   (short)0, acc0, false, false);
    acc1 = __builtin_amdgcn_wmma_f32_16x16x32_bf16(false, av, false, b1,
                                                   (short)0, acc1, false, false);
  }
  int rbase = tm + ((lane >> 4) << 3);
  int col0 = (tn2 << 5) + (lane & 15);
  int col1 = col0 + 16;
  float bc0 = bias ? bias[col0] : 0.f;
  float bc1 = bias ? bias[col1] : 0.f;
#pragma unroll
  for (int r = 0; r < 8; ++r) {
    int m = rbase + r;
    float v0 = acc0[r] + bc0;
    float v1 = acc1[r] + bc1;
    if (RELU) { v0 = fmaxf(v0, 0.f); v1 = fmaxf(v1, 0.f); }
    if (STORE_F32)  { Cf[(size_t)m * N + col0] = v0; Cf[(size_t)m * N + col1] = v1; }
    if (STORE_BF16) { Cb[(size_t)m * N + col0] = f2bf(v0); Cb[(size_t)m * N + col1] = f2bf(v1); }
  }
}

// ---- conv as GEMM over im2col + fragment-packed expert weights + BN epilogue ----
template <bool FIRST>
__global__ __launch_bounds__(256)
void k_conv_wmma(const __bf16* __restrict__ Acol, const __bf16* __restrict__ W,
                 const int* __restrict__ idx, const float* __restrict__ cb,
                 const float* __restrict__ sc, const float* __restrict__ bb,
                 const float* __restrict__ mn, const float* __restrict__ vr,
                 const float* __restrict__ resid, __bf16* __restrict__ OutB,
                 float* __restrict__ OutF, int Cin, int Cout) {
  int lane = threadIdx.x & 31;
  int wave = threadIdx.x >> 5;
  int K = Cin * 9;
  int kblocks = K >> 5;
  int tilesN2 = Cout >> 5;
  int tilesPerSample = 4 * tilesN2;          // M = 64 pixels -> 4 row tiles
  int w = blockIdx.x * 8 + wave;             // wave-uniform
  int b = w / tilesPerSample;
  int t = w - b * tilesPerSample;
  int tmIdx = t / tilesN2;
  int tn2 = t - tmIdx * tilesN2;
  int tm = tmIdx << 4;
  int e = idx[b];
  int halo = (lane >> 4) << 3;
  const __bf16* Arow = Acol + (size_t)(b * 64 + tm + (lane & 15)) * K + halo;
  const __bf16* Wb = W + (size_t)e * K * Cout;
  const __bf16* B0 = Wb + ((size_t)(2 * tn2) * kblocks * 32 + lane) * 16;
  const __bf16* B1 = B0 + (size_t)kblocks * 32 * 16;
  v8f acc0 = {}, acc1 = {};
  for (int kb = 0; kb < kblocks; ++kb) {
    __builtin_prefetch(Arow + ((kb + 4) << 5), 0, 0);
    __builtin_prefetch(B0 + (size_t)(kb + 4) * 512, 0, 0);
    __builtin_prefetch(B1 + (size_t)(kb + 4) * 512, 0, 0);
    v16bf av = load_a_frag(Arow + (kb << 5));
    v16bf b0 = *(const v16bf*)(B0 + (size_t)kb * 512);
    v16bf b1 = *(const v16bf*)(B1 + (size_t)kb * 512);
    acc0 = __builtin_amdgcn_wmma_f32_16x16x32_bf16(false, av, false, b0,
                                                   (short)0, acc0, false, false);
    acc1 = __builtin_amdgcn_wmma_f32_16x16x32_bf16(false, av, false, b1,
                                                   (short)0, acc1, false, false);
  }
  int col0 = (tn2 << 5) + (lane & 15);
  int col1 = col0 + 16;
  int co0 = e * Cout + col0, co1 = e * Cout + col1;
  float inv0 = sc[co0] * rsqrtf(vr[co0] + BN_EPS);
  float inv1 = sc[co1] * rsqrtf(vr[co1] + BN_EPS);
  float a0 = cb[co0] - mn[co0], a1 = cb[co1] - mn[co1];
  float beta0 = bb[co0], beta1 = bb[co1];
  int rbase = tm + ((lane >> 4) << 3);
#pragma unroll
  for (int r = 0; r < 8; ++r) {
    int m = rbase + r;
    float v0 = (acc0[r] + a0) * inv0 + beta0;
    float v1 = (acc1[r] + a1) * inv1 + beta1;
    size_t o0 = ((size_t)b * 64 + m) * Cout + col0;
    size_t o1 = ((size_t)b * 64 + m) * Cout + col1;
    if (FIRST) {
      OutB[o0] = f2bf(fmaxf(v0, 0.f));
      OutB[o1] = f2bf(fmaxf(v1, 0.f));
    } else {
      OutF[o0] = fmaxf(v0 + resid[o0], 0.f);
      OutF[o1] = fmaxf(v1 + resid[o1], 0.f);
    }
  }
}

// ---------------- host side ----------------
extern "C" void kernel_launch(void* const* d_in, const int* in_sizes, int n_in,
                              void* d_out, int out_size, void* d_ws, size_t ws_size,
                              hipStream_t stream) {
  (void)in_sizes; (void)n_in; (void)out_size; (void)ws_size;
  static const int Es[3] = {1, 2, 4};

  // workspace carve-up (256B aligned)
  char* base = (char*)d_ws;
  size_t off = 0;
  auto take = [&](size_t bytes) { char* p = base + off; off = (off + bytes + 255) & ~(size_t)255; return (void*)p; };
  float*  XP0  = (float*) take((size_t)1024 * 512 * 4);
  float*  XP1  = (float*) take((size_t)1024 * 512 * 4);
  __bf16* XB   = (__bf16*)take((size_t)1024 * 512 * 2);
  __bf16* W1B  = (__bf16*)take((size_t)512 * 2048 * 2);
  __bf16* HB   = (__bf16*)take((size_t)1024 * 2048 * 2);
  __bf16* W2B  = (__bf16*)take((size_t)2048 * 512 * 2);
  float*  MLPF = (float*) take((size_t)1024 * 512 * 4);
  __bf16* CW1B = (__bf16*)take((size_t)4 * 4608 * 256 * 2);
  __bf16* CW2B = (__bf16*)take((size_t)4 * 2304 * 512 * 2);
  __bf16* AC1  = (__bf16*)take((size_t)1024 * 4608 * 2);   // im2col for conv1
  __bf16* AC2  = (__bf16*)take((size_t)1024 * 2304 * 2);   // im2col for conv2
  __bf16* HC   = (__bf16*)take((size_t)16 * 64 * 256 * 2);
  int*    IDX  = (int*)   take((size_t)16 * 4);

  const int nxe = 16 * 512 * 64;
  k_nchw_to_pix<<<(nxe + 255) / 256, 256, 0, stream>>>((const float*)d_in[0], XP0, nxe);

  float* xpA = XP0;
  float* xpB = XP1;
  for (int l = 0; l < 3; ++l) {
    const int E = Es[l];
    const float* w1  = (const float*)d_in[1 + 4 * l + 0];
    const float* b1  = (const float*)d_in[1 + 4 * l + 1];
    const float* w2  = (const float*)d_in[1 + 4 * l + 2];
    const float* b2  = (const float*)d_in[1 + 4 * l + 3];
    const float* gw1 = (const float*)d_in[13 + 4 * l + 0];
    const float* gb1 = (const float*)d_in[13 + 4 * l + 1];
    const float* gw2 = (const float*)d_in[13 + 4 * l + 2];
    const float* gb2 = (const float*)d_in[13 + 4 * l + 3];
    const float* c1w = (const float*)d_in[25 + 8 * l + 0];
    const float* c1b = (const float*)d_in[25 + 8 * l + 1];
    const float* s1  = (const float*)d_in[25 + 8 * l + 2];
    const float* bb1 = (const float*)d_in[25 + 8 * l + 3];
    const float* c2w = (const float*)d_in[25 + 8 * l + 4];
    const float* c2b = (const float*)d_in[25 + 8 * l + 5];
    const float* s2  = (const float*)d_in[25 + 8 * l + 6];
    const float* bb2 = (const float*)d_in[25 + 8 * l + 7];
    const float* m1  = (const float*)d_in[49 + 4 * l + 0];
    const float* v1  = (const float*)d_in[49 + 4 * l + 1];
    const float* m2  = (const float*)d_in[49 + 4 * l + 2];
    const float* v2  = (const float*)d_in[49 + 4 * l + 3];

    k_gate<<<16, 256, 0, stream>>>(xpA, gw1, gb1, gw2, gb2, E, IDX);

    int n;
    n = 1024 * 512;  k_cast_bf16<<<(n + 255) / 256, 256, 0, stream>>>(xpA, XB, n);
    n = 512 * 2048;  k_pack_b_frag<<<(n + 255) / 256, 256, 0, stream>>>(w1, W1B, 2048, 512, n);
    n = 2048 * 512;  k_pack_b_frag<<<(n + 255) / 256, 256, 0, stream>>>(w2, W2B, 512, 2048, n);
    n = E * 256 * 512 * 9; k_pack_conv_frag<<<(n + 255) / 256, 256, 0, stream>>>(c1w, CW1B, 256, 512, n);
    n = E * 512 * 256 * 9; k_pack_conv_frag<<<(n + 255) / 256, 256, 0, stream>>>(c2w, CW2B, 512, 256, n);

    // fc1: [1024,512]x[512,2048], ReLU, bf16 out (64 Mt * 64 Nt2 waves / 8)
    k_gemm_wmma<true, false, true><<<(64 * 64) / 8, 256, 0, stream>>>(
        XB, W1B, b1, nullptr, HB, 1024, 2048, 512);
    // fc2: [1024,2048]x[2048,512] -> mlp_out fp32
    k_gemm_wmma<false, true, false><<<(64 * 16) / 8, 256, 0, stream>>>(
        HB, W2B, b2, MLPF, nullptr, 1024, 512, 2048);

    // im2col for conv1 from mlp_out (fp32 -> bf16)
    n = 1024 * 4608; k_im2col_f32<<<(n + 255) / 256, 256, 0, stream>>>(MLPF, AC1, 512, n);
    // conv1: 512->256, K=4608, bias+BN+ReLU -> bf16 (16 samples * 32 wave-tiles / 8)
    k_conv_wmma<true><<<(16 * 32) / 8, 256, 0, stream>>>(
        AC1, CW1B, IDX, c1b, s1, bb1, m1, v1, nullptr, HC, nullptr, 512, 256);
    // im2col for conv2 from hidden (bf16)
    n = 1024 * 2304; k_im2col_bf16<<<(n + 255) / 256, 256, 0, stream>>>(HC, AC2, 256, n);
    // conv2: 256->512, K=2304, bias+BN + residual(mlp_out) + ReLU -> fp32
    k_conv_wmma<false><<<(16 * 64) / 8, 256, 0, stream>>>(
        AC2, CW2B, IDX, c2b, s2, bb2, m2, v2, MLPF, nullptr, xpB, 256, 512);

    float* tmp = xpA; xpA = xpB; xpB = tmp;
  }

  k_pix_to_nchw<<<(nxe + 255) / 256, 256, 0, stream>>>(xpA, (float*)d_out, nxe);
}
